// pit_26439818674935
// MI455X (gfx1250) — matrix operand order repaired and checked by reading, not compiled, not run on hardware
//
#include <hip/hip_runtime.h>
#include <hip/hip_bf16.h>
#include <math.h>

typedef __attribute__((ext_vector_type(16))) _Float16 v16h;
typedef __attribute__((ext_vector_type(8)))  _Float16 v8h;
typedef __attribute__((ext_vector_type(8)))  float    v8f;

#define FMAXV 3.402823466e+38f

// ---------------------------------------------------------------------------
// Attention: one block per (b,h,row). Computes scaled squared distances,
// exact per-row quantile threshold (radix select on float bits), and stores
// unnormalized softmax numerators exp(sd_min - sd) as f16 plus f32 row sum Z.
// ---------------------------------------------------------------------------
__global__ void k_attn(const float* __restrict__ mesh_a, const float* __restrict__ mesh_b,
                       const float* __restrict__ lmda, _Float16* __restrict__ E,
                       float* __restrict__ Z, int Na, int Mb, int H,
                       int k0, float frac)
{
  extern __shared__ float smem[];
  float*    sd    = smem;           // Mb floats
  float*    red   = smem + Mb;      // 256 floats
  unsigned* hist  = (unsigned*)(red + 256);  // 256
  unsigned* bcast = hist + 256;              // 2

  int tid = threadIdx.x;
  long z  = blockIdx.x;
  int n   = (int)(z % Na);
  int bh  = (int)(z / Na);
  int h   = bh % H;
  int b   = bh / H;

  const float cst = 0.25f * 3.14159265358979323846f * (1.0f - 1e-7f);
  float scale = tanf(cst * (1.0f + sinf(lmda[h])));

  float xa = mesh_a[((long)b * Na + n) * 2 + 0];
  float ya = mesh_a[((long)b * Na + n) * 2 + 1];
  const float* mbp = mesh_b + (long)b * Mb * 2;

  float lmin = FMAXV;
  for (int j = tid; j < Mb; j += blockDim.x) {
    float dx = xa - mbp[2 * j + 0];
    float dy = ya - mbp[2 * j + 1];
    float s  = (dx * dx + dy * dy) * scale;
    sd[j] = s;
    lmin  = fminf(lmin, s);
  }
  red[tid] = lmin;
  __syncthreads();
  for (int st = blockDim.x / 2; st > 0; st >>= 1) {
    if (tid < st) red[tid] = fminf(red[tid], red[tid + st]);
    __syncthreads();
  }
  float sdmin = red[0];
  __syncthreads();

  float thr = FMAXV;
  if (k0 >= 0) {
    unsigned v2[2];
    for (int t = 0; t < 2; ++t) {
      int rem = k0 + t;
      unsigned prefix = 0;
      for (int shift = 24; shift >= 0; shift -= 8) {
        if (tid < 256) hist[tid] = 0u;
        __syncthreads();
        unsigned pmask = (shift == 24) ? 0u : (0xFFFFFFFFu << (shift + 8));
        for (int j = tid; j < Mb; j += blockDim.x) {
          unsigned u = __float_as_uint(sd[j]);
          if ((u & pmask) == prefix) atomicAdd(&hist[(u >> shift) & 255u], 1u);
        }
        __syncthreads();
        if (tid == 0) {
          int cum = 0; unsigned sel = 255;
          for (int bin = 0; bin < 256; ++bin) {
            int hc = (int)hist[bin];
            if (cum + hc > rem) { sel = (unsigned)bin; break; }
            cum += hc;
          }
          bcast[0] = sel; bcast[1] = (unsigned)(rem - cum);
        }
        __syncthreads();
        prefix |= (bcast[0] << shift);
        rem = (int)bcast[1];
        __syncthreads();
      }
      v2[t] = prefix;
    }
    float lo = __uint_as_float(v2[0]);
    float hi = __uint_as_float(v2[1]);
    thr = lo + frac * (hi - lo);
  }

  _Float16* Erow = E + ((long)bh * Na + n) * Mb;
  float lsum = 0.f;
  for (int j = tid; j < Mb; j += blockDim.x) {
    float s = sd[j];
    float e = (s <= thr) ? expf(sdmin - s) : 0.f;
    Erow[j] = (_Float16)e;
    lsum += e;
  }
  red[tid] = lsum;
  __syncthreads();
  for (int st = blockDim.x / 2; st > 0; st >>= 1) {
    if (tid < st) red[tid] += red[tid + st];
    __syncthreads();
  }
  if (tid == 0) Z[(long)bh * Na + n] = red[0];
}

// ---------------------------------------------------------------------------
// WMMA: one wave computes NT adjacent 16x16 column tiles of a 16-row block,
// reusing the A fragment across all NT WMMAs per k-step (higher intensity).
// A row-major (lda >= K, f16); BT = B^T row-major (ldb >= K, f16).
// Fragment layouts per CDNA5 ISA 7.12.2 (wave32).
// ---------------------------------------------------------------------------
template<int NT>
__device__ __forceinline__ void gemm_tiles(const _Float16* __restrict__ A, long lda,
                                           const _Float16* __restrict__ BT, long ldb,
                                           int K, v8f acc[NT])
{
  int lane = threadIdx.x & 31;
  int r  = lane & 15;
  int hi = lane >> 4;
  const _Float16* a0p = A + (long)r * lda + 8 * hi;       // K = 8hi .. 8hi+7
  const _Float16* a1p = a0p + 16;                         // K = 16+8hi .. +7
  const _Float16* bp  = BT + (long)r * ldb + 8 * hi;      // col r, K = 8hi..+7, 16+8hi..+7
  for (int k = 0; k < K; k += 32) {
    union { v16h v; v8h p[2]; } au;
    au.p[0] = *(const v8h*)(a0p + k);
    au.p[1] = *(const v8h*)(a1p + k);
#pragma unroll
    for (int t = 0; t < NT; ++t) {
      union { v16h v; v8h p[2]; } bu;
      const _Float16* bpt = bp + (long)t * 16 * ldb;
      bu.p[0] = *(const v8h*)(bpt + k);
      bu.p[1] = *(const v8h*)(bpt + k + 16);
      acc[t] = __builtin_amdgcn_wmma_f32_16x16x32_f16(false, au.v, false, bu.v,
                                                      (short)0, acc[t], false, false);
    }
  }
}

// Conv: out[b,n, ycoloff + h*hstride + d] = (1/Z[b,h,n]) * sum_j E[b,h,n,j]*VT[b,d,j]
template<int NT>
__global__ void k_conv(const _Float16* __restrict__ E, const float* __restrict__ Z,
                       const _Float16* __restrict__ VT, _Float16* __restrict__ Y16,
                       long ldy, long ycoloff, int N, int M, int Dp, int Dvalid,
                       int H, int hstride)
{
  int lane = threadIdx.x;
  int z = blockIdx.z; int h = z % H; int b = z / H;
  int n0 = blockIdx.x * 16;
  int d0 = blockIdx.y * 16 * NT;
  const _Float16* At = E  + ((long)z * N + n0) * M;
  const _Float16* Bt = VT + ((long)b * Dp + d0) * M;
  v8f acc[NT];
#pragma unroll
  for (int t = 0; t < NT; ++t) acc[t] = (v8f){0.f,0.f,0.f,0.f,0.f,0.f,0.f,0.f};
  gemm_tiles<NT>(At, M, Bt, M, M, acc);
  int col = lane & 15, hi = lane >> 4;
#pragma unroll
  for (int t = 0; t < NT; ++t) {
    int d = d0 + t * 16 + col;
    if (d < Dvalid) {
      for (int i = 0; i < 8; ++i) {
        int n = n0 + i + 8 * hi;
        float rz = Z[(long)z * N + n];
        float val = acc[t][i] / rz;
        Y16[((long)b * N + n) * ldy + ycoloff + (long)h * hstride + d] = (_Float16)val;
      }
    }
  }
}

// GEMM + bias + optional exact GELU; writes f32 and/or f16.
template<int NT>
__global__ void k_gemm(const _Float16* __restrict__ A, long lda,
                       const _Float16* __restrict__ BT, long ldb,
                       const float* __restrict__ bias, int Cvalid, int actgelu,
                       float* __restrict__ Yf, long ldyf,
                       _Float16* __restrict__ Y16, long ldy16, int K)
{
  int lane = threadIdx.x;
  int r0 = blockIdx.x * 16;
  int c0 = blockIdx.y * 16 * NT;
  v8f acc[NT];
#pragma unroll
  for (int t = 0; t < NT; ++t) acc[t] = (v8f){0.f,0.f,0.f,0.f,0.f,0.f,0.f,0.f};
  gemm_tiles<NT>(A + (long)r0 * lda, lda, BT + (long)c0 * ldb, ldb, K, acc);
  int col = lane & 15, hi = lane >> 4;
#pragma unroll
  for (int t = 0; t < NT; ++t) {
    int cc = c0 + t * 16 + col;
    if (cc < Cvalid) {
      float bv = bias ? bias[cc] : 0.f;
      for (int i = 0; i < 8; ++i) {
        int rr = r0 + i + 8 * hi;
        float y = acc[t][i] + bv;
        if (actgelu) y = 0.5f * y * (1.0f + erff(y * 0.70710678118654752f));
        if (Yf)  Yf[(long)rr * ldyf + cc] = y;
        if (Y16) Y16[(long)rr * ldy16 + cc] = (_Float16)y;
      }
    }
  }
}

// ---------------------------------------------------------------------------
// Small prep kernels
// ---------------------------------------------------------------------------
__global__ void k_transpose_cast(const float* __restrict__ src, _Float16* __restrict__ dst,
                                 int B_, int N, int D, int Dp)
{ // src (B,N,D) f32 -> dst (B,Dp,N) f16 (rows >= D zero)
  long idx = (long)blockIdx.x * blockDim.x + threadIdx.x;
  long total = (long)B_ * Dp * N;
  if (idx >= total) return;
  int n = (int)(idx % N); long t = idx / N;
  int d = (int)(t % Dp);  int b = (int)(t / Dp);
  float v = (d < D) ? src[((long)b * N + n) * D + d] : 0.f;
  dst[idx] = (_Float16)v;
}

__global__ void k_copy_cast(const float* __restrict__ src, _Float16* __restrict__ dst,
                            int rows, int scols, long dld, long coff)
{
  long idx = (long)blockIdx.x * blockDim.x + threadIdx.x;
  long total = (long)rows * scols;
  if (idx >= total) return;
  long r = idx / scols, c = idx % scols;
  dst[r * dld + coff + c] = (_Float16)src[idx];
}

__global__ void k_wt(const float* __restrict__ w, _Float16* __restrict__ wt,
                     int K, int C, int Kp, int Cp)
{ // w (K,C) f32 -> wt (Cp,Kp) f16 transposed, zero padded
  long idx = (long)blockIdx.x * blockDim.x + threadIdx.x;
  long total = (long)Cp * Kp;
  if (idx >= total) return;
  int c = (int)(idx / Kp), k = (int)(idx % Kp);
  wt[idx] = (c < C && k < K) ? (_Float16)w[(long)k * C + c] : (_Float16)0.f;
}

__global__ void k_build_uT(const float* __restrict__ func_in, const float* __restrict__ mesh_in,
                           _Float16* __restrict__ uT, int B_, int L)
{ // uT (B,16,L): rows 0..2 func, 3..4 mesh, rest zero
  long idx = (long)blockIdx.x * blockDim.x + threadIdx.x;
  long total = (long)B_ * 16 * L;
  if (idx >= total) return;
  int j = (int)(idx % L); long t = idx / L;
  int d = (int)(t % 16);  int b = (int)(t / 16);
  float v = 0.f;
  if (d < 3)      v = func_in[((long)b * L + j) * 3 + d];
  else if (d < 5) v = mesh_in[((long)b * L + j) * 2 + (d - 3)];
  uT[idx] = (_Float16)v;
}

// ---------------------------------------------------------------------------
extern "C" void kernel_launch(void* const* d_in, const int* in_sizes, int n_in,
                              void* d_out, int out_size, void* d_ws, size_t ws_size,
                              hipStream_t stream) {
  const float* mesh_in   = (const float*)d_in[0];
  const float* func_in   = (const float*)d_in[1];
  const float* mesh_ltt  = (const float*)d_in[2];
  const float* mesh_out  = (const float*)d_in[3];
  const float* lmda_down = (const float*)d_in[4];
  const float* en_w1     = (const float*)d_in[5];
  const float* en_b1     = (const float*)d_in[6];
  const float* en_w2     = (const float*)d_in[7];
  const float* en_b2     = (const float*)d_in[8];
  const float* lmda_blk  = (const float*)d_in[9];
  const float* blk_w1    = (const float*)d_in[10];
  const float* blk_b1    = (const float*)d_in[11];
  const float* blk_w2    = (const float*)d_in[12];
  const float* blk_b2    = (const float*)d_in[13];
  const float* lmda_up   = (const float*)d_in[14];
  const float* de_w1     = (const float*)d_in[15];
  const float* de_b1     = (const float*)d_in[16];
  const float* de_w2     = (const float*)d_in[17];
  const float* de_b2     = (const float*)d_in[18];

  const int B = 4, Lin = 2048, Lltt = 1024, Lout = 2048, H = 4, hid = 128;

  char* p = (char*)d_ws;
  auto carve = [&](size_t bytes) { char* q = p; p += (bytes + 255) & ~(size_t)255; return q; };
  _Float16* E     = (_Float16*)carve((size_t)B * H * Lout * Lltt * 2); // 64 MB (covers enc too)
  float*    Zbuf  = (float*)   carve((size_t)B * H * Lout * 4);
  _Float16* uT    = (_Float16*)carve((size_t)B * 16 * Lin * 2);
  _Float16* fT    = (_Float16*)carve((size_t)B * hid * Lltt * 2);
  float*    f32b  = (float*)   carve((size_t)B * Lltt * hid * 4);
  _Float16* X16   = (_Float16*)carve((size_t)B * Lout * 512 * 2);      // max stage
  _Float16* Y1    = (_Float16*)carve((size_t)B * Lout * hid * 2);
  _Float16* enw1T = (_Float16*)carve(128 * 32 * 2);
  _Float16* enw2T = (_Float16*)carve(128 * 128 * 2);
  _Float16* bw1T  = (_Float16*)carve((size_t)4 * 128 * 640 * 2);
  _Float16* bw2T  = (_Float16*)carve((size_t)4 * 128 * 128 * 2);
  _Float16* dw1T  = (_Float16*)carve(128 * 512 * 2);
  _Float16* dw2T  = (_Float16*)carve(16 * 128 * 2);

  auto nb = [](long total, int t) { return (unsigned)((total + t - 1) / t); };
  const int T = 256;

  // ---- weight prep (f16, transposed, padded) ----
  k_wt<<<nb(128L * 32, T), T, 0, stream>>>(en_w1, enw1T, 20, 128, 32, 128);
  k_wt<<<nb(128L * 128, T), T, 0, stream>>>(en_w2, enw2T, 128, 128, 128, 128);
  for (int i = 0; i < 4; ++i) {
    k_wt<<<nb(128L * 640, T), T, 0, stream>>>(blk_w1 + (long)i * 640 * 128,
                                              bw1T + (long)i * 128 * 640, 640, 128, 640, 128);
    k_wt<<<nb(128L * 128, T), T, 0, stream>>>(blk_w2 + (long)i * 128 * 128,
                                              bw2T + (long)i * 128 * 128, 128, 128, 128, 128);
  }
  k_wt<<<nb(128L * 512, T), T, 0, stream>>>(de_w1, dw1T, 512, 128, 512, 128);
  k_wt<<<nb(128L * 16, T), T, 0, stream>>>(de_w2, dw2T, 128, 1, 128, 16);

  // quantile positions (jnp.quantile 'linear': pos = q*(M-1))
  double posE = 0.05 * (Lin - 1);  int k0E = (int)posE; float frE = (float)(posE - k0E);
  double posD = 0.05 * (Lltt - 1); int k0D = (int)posD; float frD = (float)(posD - k0D);

  size_t shEnc = (size_t)(Lin  + 256 + 256 + 8) * 4;
  size_t shLtt = (size_t)(Lltt + 256 + 256 + 8) * 4;

  // ================= encoder =================
  k_build_uT<<<nb((long)B * 16 * Lin, T), T, 0, stream>>>(func_in, mesh_in, uT, B, Lin);
  k_attn<<<B * H * Lltt, T, shEnc, stream>>>(mesh_ltt, mesh_in, lmda_down, E, Zbuf,
                                             Lltt, Lin, H, k0E, frE);
  hipMemsetAsync(X16, 0, (size_t)B * Lltt * 32 * 2, stream);  // pad cols of (4096,32)
  {
    dim3 g(Lltt / 16, 1, B * H);
    k_conv<1><<<g, 32, 0, stream>>>(E, Zbuf, uT, X16, 32, 0, Lltt, Lin, 16, 5, H, 5);
    dim3 gm(B * Lltt / 16, hid / 64);
    k_gemm<4><<<gm, 32, 0, stream>>>(X16, 32, enw1T, 32, en_b1, 128, 1,
                                     nullptr, 0, Y1, 128, 32);
    k_gemm<4><<<gm, 32, 0, stream>>>(Y1, 128, enw2T, 128, en_b2, 128, 1,
                                     f32b, 128, nullptr, 0, 128);
  }

  // ================= processor blocks =================
  for (int i = 0; i < 4; ++i) {
    k_attn<<<B * H * Lltt, T, shLtt, stream>>>(mesh_ltt, mesh_ltt, lmda_blk + i * H,
                                               E, Zbuf, Lltt, Lltt, H, -1, 0.f);
    k_transpose_cast<<<nb((long)B * hid * Lltt, T), T, 0, stream>>>(f32b, fT, B, Lltt, hid, hid);
    k_copy_cast<<<nb((long)B * Lltt * hid, T), T, 0, stream>>>(f32b, X16, B * Lltt, hid, 640, 0);
    dim3 gc(Lltt / 16, hid / 64, B * H);
    k_conv<4><<<gc, 32, 0, stream>>>(E, Zbuf, fT, X16, 640, 128, Lltt, Lltt, hid, hid, H, hid);
    dim3 gm(B * Lltt / 16, hid / 64);
    k_gemm<4><<<gm, 32, 0, stream>>>(X16, 640, bw1T + (long)i * 128 * 640, 640,
                                     blk_b1 + i * 128, 128, 1, nullptr, 0, Y1, 128, 640);
    k_gemm<4><<<gm, 32, 0, stream>>>(Y1, 128, bw2T + (long)i * 128 * 128, 128,
                                     blk_b2 + i * 128, 128, 1, f32b, 128, nullptr, 0, 128);
  }

  // ================= decoder =================
  k_attn<<<B * H * Lout, T, shLtt, stream>>>(mesh_out, mesh_ltt, lmda_up, E, Zbuf,
                                             Lout, Lltt, H, k0D, frD);
  k_transpose_cast<<<nb((long)B * hid * Lltt, T), T, 0, stream>>>(f32b, fT, B, Lltt, hid, hid);
  {
    dim3 gc(Lout / 16, hid / 64, B * H);
    k_conv<4><<<gc, 32, 0, stream>>>(E, Zbuf, fT, X16, 512, 0, Lout, Lltt, hid, hid, H, hid);
    dim3 gm1(B * Lout / 16, hid / 64);
    k_gemm<4><<<gm1, 32, 0, stream>>>(X16, 512, dw1T, 512, de_b1, 128, 1,
                                      nullptr, 0, Y1, 128, 512);
    dim3 gm2(B * Lout / 16, 1);
    k_gemm<1><<<gm2, 32, 0, stream>>>(Y1, 128, dw2T, 128, de_b2, 1, 0,
                                      (float*)d_out, 1, nullptr, 0, 128);
  }
}
